// mLSTMLayer_38156489458049
// MI455X (gfx1250) — matrix operand order repaired and checked by reading, not compiled, not run on hardware
//
#include <hip/hip_runtime.h>
#include <hip/hip_bf16.h>

typedef __bf16 bf16_t;
typedef __attribute__((ext_vector_type(16))) __bf16 v16bf;
typedef __attribute__((ext_vector_type(8)))  float  v8f;

#define B_   64
#define S_   512
#define IN_  512
#define H_   1024
#define NWG  64
#define TPB  128

// LDS staging: weight rows padded +16B so the 16 lane rows hit distinct banks.
#define LDS_STRIDE_IN (IN_ * 2 + 16)   // 1040
#define LDS_STRIDE_H  (H_ * 2 + 16)    // 2064
#define OFF_WMX 0
#define OFF_WMH (OFF_WMX + 16 * LDS_STRIDE_IN)          // 16640
#define OFF_WX  (OFF_WMH + 16 * LDS_STRIDE_H)           // 49664
#define OFF_WH  (OFF_WX + 64 * LDS_STRIDE_IN)           // 116224
#define LDS_TOTAL (OFF_WH + 64 * LDS_STRIDE_H)          // 248320 <= 320KB/WGP

// ---------------- WMMA helpers (CDNA5 wave32, 16x16x32 bf16 -> f32) ----------

__device__ __forceinline__ v8f wmma_bf16(v8f acc, v16bf a, v16bf b) {
  return __builtin_amdgcn_wmma_f32_16x16x32_bf16(false, a, false, b, (short)0, acc,
                                                 false, false);
}

// A-matrix 16x32 bf16: lane<16 -> row M=lane, K = {c*32+0..7, c*32+16..23};
// lane>=16 -> row M=lane-16, K = {c*32+8..15, c*32+24..31}. Two 16B chunks at
// p and p+32 (p already includes the per-lane +0/+16 byte shift).
__device__ __forceinline__ v16bf load_a16(const char* p) {
  uint4 lo = *(const uint4*)(p);
  uint4 hi = *(const uint4*)(p + 32);
  v16bf r;
  ((uint4*)&r)[0] = lo;
  ((uint4*)&r)[1] = hi;
  return r;
}

// B-matrix 32x16 bf16 from LDS: lane<16 -> col N=lane, K = {c*32+0..7, c*32+16..23};
// lane>=16 -> col N=lane-16, K = {c*32+8..15, c*32+24..31}. Two 16B chunks at p and p+32.
__device__ __forceinline__ v16bf load_b_lds(const char* p) {
  uint4 lo = *(const uint4*)(p);
  uint4 hi = *(const uint4*)(p + 32);
  v16bf r;
  ((uint4*)&r)[0] = lo;
  ((uint4*)&r)[1] = hi;
  return r;
}

__device__ __forceinline__ float sigmoidf_(float x) {
  return 1.0f / (1.0f + __expf(-x));
}
__device__ __forceinline__ float tanhf_(float x) {
  x = fminf(15.0f, fmaxf(-15.0f, x));
  float e = __expf(2.0f * x);
  return (e - 1.0f) / (e + 1.0f);
}

// Monotonic-count software grid barrier across NWG workgroups.
__device__ __forceinline__ void grid_barrier(unsigned* bar, unsigned target) {
  __syncthreads();
  if (threadIdx.x == 0) {
    __threadfence();              // release: make m/h stores visible device-wide
    atomicAdd(bar, 1u);
    while (atomicAdd(bar, 0u) < target) {
      __builtin_amdgcn_s_sleep(2);
    }
  }
  __syncthreads();
  __threadfence();                // acquire before reading peers' m/h
}

// Stage weight rows (bf16, row-major (N,K)) into padded LDS rows.
// Local row r maps to global row (r>>4)*H_ + (r&15) relative to gbase.
__device__ __forceinline__ void stage(char* dst, const bf16_t* gbase, int nrows,
                                      int kelems, int ldstride) {
  int cpr = (kelems * 2) >> 4;  // 16B chunks per row
  int total = nrows * cpr;
  for (int i = threadIdx.x; i < total; i += TPB) {
    int r = i / cpr;
    int c = i - r * cpr;
    size_t goff = ((size_t)(r >> 4) * H_ + (size_t)(r & 15)) * (size_t)kelems;
    *(uint4*)(dst + r * ldstride + c * 16) =
        *(const uint4*)((const char*)(gbase + goff) + c * 16);
  }
}

// ---------------- small utility kernels -------------------------------------

__global__ void cvt_f32_bf16(const float* __restrict__ in, bf16_t* __restrict__ out,
                             int n) {
  int stride = gridDim.x * blockDim.x;
  for (int i = blockIdx.x * blockDim.x + threadIdx.x; i < n; i += stride)
    out[i] = (bf16_t)in[i];
}

__global__ void init_state(bf16_t* __restrict__ h_buf, unsigned* __restrict__ bar) {
  int i = blockIdx.x * blockDim.x + threadIdx.x;
  if (i == 0) bar[0] = 0u;
  int n = B_ * H_;
  int stride = gridDim.x * blockDim.x;
  for (int j = i; j < n; j += stride) h_buf[j] = (bf16_t)0.0f;
}

// ---------------- persistent mLSTM scan --------------------------------------
// Grid: 64 WGs (one per hidden tile) x 128 threads (4 waves = 4 batch tiles).
// All weights for this WG's hidden tile live in LDS for the whole scan.
// Per step:
//   Phase A: m_tile = (x_t @ wmx^T) * (h @ wmh^T)        -> bf16 m_buf (global)
//   [grid barrier]
//   Phase B: 4 gate tiles = bias + x_t @ wx_g^T + m @ wh_g^T ; update h,c;
//            write outs f32; write h bf16 to shared h_buf
//   [grid barrier]

__global__ void __launch_bounds__(TPB, 1)
mlstm_scan_kernel(const bf16_t* __restrict__ xb,
                  const float*  __restrict__ mask,
                  const bf16_t* __restrict__ wmxb,
                  const bf16_t* __restrict__ wmhb,
                  const bf16_t* __restrict__ wxb,
                  const bf16_t* __restrict__ whb,
                  const float*  __restrict__ bias,
                  bf16_t* __restrict__ m_buf,
                  bf16_t* __restrict__ h_buf,
                  unsigned* __restrict__ bar,
                  float* __restrict__ outs,
                  float* __restrict__ hT,
                  float* __restrict__ cT) {
  extern __shared__ char smem[];

  const int lane  = threadIdx.x & 31;
  const int Mtile = threadIdx.x >> 5;   // 0..3: batch tile of 16 rows
  const int ntile = blockIdx.x;         // 0..63: hidden tile of 16 cols

  const int l15  = lane & 15;
  const int a0   = (lane >= 16) ? 16 : 0;   // A byte shift within 64B k-step
  const int b0   = (lane >= 16) ? 16 : 0;   // B byte shift within 64B k-step
  const int n    = ntile * 16 + l15;        // hidden column owned (C layout)
  const int arow = Mtile * 16 + l15;        // batch row loaded for A-matrix
  const int crow = Mtile * 16 + ((lane >= 16) ? 8 : 0);  // C-layout row base

  // ---- stage all weights for this hidden tile into LDS (once) ----
  stage(smem + OFF_WMX, wmxb + (size_t)(ntile * 16) * IN_, 16, IN_, LDS_STRIDE_IN);
  stage(smem + OFF_WMH, wmhb + (size_t)(ntile * 16) * H_, 16, H_, LDS_STRIDE_H);
  stage(smem + OFF_WX,  wxb  + (size_t)(ntile * 16) * IN_, 64, IN_, LDS_STRIDE_IN);
  stage(smem + OFF_WH,  whb  + (size_t)(ntile * 16) * H_, 64, H_, LDS_STRIDE_H);
  __syncthreads();

  // per-lane LDS row bases (N = l15 within the tile)
  const char* wmx_l = smem + OFF_WMX + l15 * LDS_STRIDE_IN;
  const char* wmh_l = smem + OFF_WMH + l15 * LDS_STRIDE_H;
  const char* wx_l  = smem + OFF_WX  + l15 * LDS_STRIDE_IN;
  const char* wh_l  = smem + OFF_WH  + l15 * LDS_STRIDE_H;

  const char* xrow_base = (const char*)(xb + (size_t)arow * (S_ * IN_));
  const char* hrow      = (const char*)(h_buf + (size_t)arow * H_);
  const char* mrow      = (const char*)(m_buf + (size_t)arow * H_);

  const float bias_v0 = bias[0 * H_ + n];
  const float bias_v1 = bias[1 * H_ + n];
  const float bias_v2 = bias[2 * H_ + n];
  const float bias_v3 = bias[3 * H_ + n];

  v8f hreg = {};
  v8f creg = {};

  for (int t = 0; t < S_; ++t) {
    const char* xrow = xrow_base + (size_t)t * (IN_ * 2);
    __builtin_prefetch(xrow + IN_ * 2, 0, 1);  // next step's x row -> L2

    // ---- Phase A: m = (x_t @ wmx^T) * (h @ wmh^T) ----
    v8f accX = {};
#pragma unroll
    for (int c = 0; c < IN_ / 32; ++c)
      accX = wmma_bf16(accX, load_a16(xrow + c * 64 + a0),
                             load_b_lds(wmx_l + c * 64 + b0));
    v8f accH = {};
#pragma unroll
    for (int c = 0; c < H_ / 32; ++c)
      accH = wmma_bf16(accH, load_a16(hrow + c * 64 + a0),
                             load_b_lds(wmh_l + c * 64 + b0));
    v8f mval = accX * accH;
#pragma unroll
    for (int r = 0; r < 8; ++r)
      m_buf[(size_t)(crow + r) * H_ + n] = (bf16_t)mval[r];

    grid_barrier(bar, (unsigned)NWG * (unsigned)(2 * t + 1));

    // ---- Phase B: shared A-tile per k-step feeds all 4 gate WMMAs ----
    v8f g0, g1, g2, g3;
#pragma unroll
    for (int r = 0; r < 8; ++r) { g0[r] = bias_v0; g1[r] = bias_v1;
                                  g2[r] = bias_v2; g3[r] = bias_v3; }
#pragma unroll
    for (int c = 0; c < IN_ / 32; ++c) {
      v16bf a = load_a16(xrow + c * 64 + a0);
      g0 = wmma_bf16(g0, a, load_b_lds(wx_l + 0 * 16 * LDS_STRIDE_IN + c * 64 + b0));
      g1 = wmma_bf16(g1, a, load_b_lds(wx_l + 1 * 16 * LDS_STRIDE_IN + c * 64 + b0));
      g2 = wmma_bf16(g2, a, load_b_lds(wx_l + 2 * 16 * LDS_STRIDE_IN + c * 64 + b0));
      g3 = wmma_bf16(g3, a, load_b_lds(wx_l + 3 * 16 * LDS_STRIDE_IN + c * 64 + b0));
    }
#pragma unroll
    for (int c = 0; c < H_ / 32; ++c) {
      v16bf a = load_a16(mrow + c * 64 + a0);
      g0 = wmma_bf16(g0, a, load_b_lds(wh_l + 0 * 16 * LDS_STRIDE_H + c * 64 + b0));
      g1 = wmma_bf16(g1, a, load_b_lds(wh_l + 1 * 16 * LDS_STRIDE_H + c * 64 + b0));
      g2 = wmma_bf16(g2, a, load_b_lds(wh_l + 2 * 16 * LDS_STRIDE_H + c * 64 + b0));
      g3 = wmma_bf16(g3, a, load_b_lds(wh_l + 3 * 16 * LDS_STRIDE_H + c * 64 + b0));
    }

    // ---- elementwise state update ----
#pragma unroll
    for (int r = 0; r < 8; ++r) {
      int b = crow + r;
      float ig = sigmoidf_(g0[r]);
      float fg = sigmoidf_(g1[r]);
      float og = sigmoidf_(g2[r]);
      float gg = tanhf_(g3[r]);
      float cy = fg * creg[r] + ig * gg;
      float hy = og * tanhf_(cy);
      float sm = mask[(size_t)b * S_ + t];
      hy = sm * hy + (1.0f - sm) * hreg[r];
      cy = sm * cy + (1.0f - sm) * creg[r];
      hreg[r] = hy;
      creg[r] = cy;
      outs[((size_t)b * S_ + t) * H_ + n] = hy;
      h_buf[(size_t)b * H_ + n] = (bf16_t)hy;
    }

    grid_barrier(bar, (unsigned)NWG * (unsigned)(2 * t + 2));
  }

#pragma unroll
  for (int r = 0; r < 8; ++r) {
    hT[(size_t)(crow + r) * H_ + n] = hreg[r];
    cT[(size_t)(crow + r) * H_ + n] = creg[r];
  }
}

// ---------------- launch ------------------------------------------------------

extern "C" void kernel_launch(void* const* d_in, const int* in_sizes, int n_in,
                              void* d_out, int out_size, void* d_ws, size_t ws_size,
                              hipStream_t stream) {
  (void)in_sizes; (void)n_in; (void)out_size; (void)ws_size;

  const float* x    = (const float*)d_in[0];
  const float* mask = (const float*)d_in[1];
  const float* wmx  = (const float*)d_in[2];
  const float* wmh  = (const float*)d_in[3];
  const float* wx   = (const float*)d_in[4];
  const float* wh   = (const float*)d_in[5];
  const float* whb  = (const float*)d_in[6];

  char* ws = (char*)d_ws;
  size_t off = 0;
  auto carve = [&](size_t bytes) -> char* {
    char* p = ws + off;
    off += (bytes + 255) & ~(size_t)255;
    return p;
  };
  bf16_t*   xb    = (bf16_t*)carve((size_t)B_ * S_ * IN_ * sizeof(bf16_t));
  bf16_t*   wmxb  = (bf16_t*)carve((size_t)H_ * IN_ * sizeof(bf16_t));
  bf16_t*   wmhb  = (bf16_t*)carve((size_t)H_ * H_ * sizeof(bf16_t));
  bf16_t*   wxb   = (bf16_t*)carve((size_t)4 * H_ * IN_ * sizeof(bf16_t));
  bf16_t*   whbw  = (bf16_t*)carve((size_t)4 * H_ * H_ * sizeof(bf16_t));
  bf16_t*   m_buf = (bf16_t*)carve((size_t)B_ * H_ * sizeof(bf16_t));
  bf16_t*   h_buf = (bf16_t*)carve((size_t)B_ * H_ * sizeof(bf16_t));
  unsigned* bar   = (unsigned*)carve(256);

  auto cvt = [&](const float* in, bf16_t* out, int n) {
    int blocks = (n + 256 * 16 - 1) / (256 * 16);
    if (blocks < 1) blocks = 1;
    hipLaunchKernelGGL(cvt_f32_bf16, dim3(blocks), dim3(256), 0, stream, in, out, n);
  };
  cvt(x,   xb,   B_ * S_ * IN_);
  cvt(wmx, wmxb, H_ * IN_);
  cvt(wmh, wmhb, H_ * H_);
  cvt(wx,  wxb,  4 * H_ * IN_);
  cvt(wh,  whbw, 4 * H_ * H_);

  hipLaunchKernelGGL(init_state, dim3(64), dim3(256), 0, stream, h_buf, bar);

  float* outs = (float*)d_out;
  float* hT   = outs + (size_t)B_ * S_ * H_;
  float* cT   = hT + (size_t)B_ * H_;

  (void)hipFuncSetAttribute((const void*)mlstm_scan_kernel,
                            hipFuncAttributeMaxDynamicSharedMemorySize, LDS_TOTAL);

  hipLaunchKernelGGL(mlstm_scan_kernel, dim3(NWG), dim3(TPB), LDS_TOTAL, stream,
                     xb, mask, wmxb, wmhb, wxb, whbw, whb,
                     m_buf, h_buf, bar, outs, hT, cT);
}
